// StackedLSTM_68985764708864
// MI455X (gfx1250) — matrix operand order repaired and checked by reading, not compiled, not run on hardware
//
#include <hip/hip_runtime.h>
#include <stdint.h>

typedef __bf16 bhalf;
typedef __attribute__((ext_vector_type(16))) __bf16 v16bf;
typedef __attribute__((ext_vector_type(8)))  float  v8f;

union Frag { v16bf v; uint4 q[2]; };

__device__ __forceinline__ bhalf f2bf(float x) {
  union { float f; uint32_t u; } a; a.f = x;
  uint32_t r = a.u + 0x7FFFu + ((a.u >> 16) & 1u);   // round-to-nearest-even
  uint16_t s = (uint16_t)(r >> 16);
  bhalf o; __builtin_memcpy(&o, &s, 2); return o;
}

__device__ __forceinline__ float sigm(float x) { return 1.0f / (1.0f + __expf(-x)); }

// ---------------- elementwise helpers ----------------
__global__ void k_cvt_bf16(const float* __restrict__ src, bhalf* __restrict__ dst, int n) {
  int i = blockIdx.x * blockDim.x + threadIdx.x;
  if (i < n) dst[i] = f2bf(src[i]);
}

// src: [K, N] row-major fp32  ->  dst: [N, K] row-major bf16
__global__ void k_transpose_bf16(const float* __restrict__ src, bhalf* __restrict__ dst,
                                 int K, int N) {
  int i = blockIdx.x * blockDim.x + threadIdx.x;
  if (i < K * N) {
    int k = i / N, n = i - k * N;
    dst[(size_t)n * K + k] = f2bf(src[i]);
  }
}

__global__ void k_zero_u32(uint32_t* __restrict__ p, int n) {
  int i = blockIdx.x * blockDim.x + threadIdx.x;
  if (i < n) p[i] = 0u;
}

__global__ void k_copy_f32(const float* __restrict__ s, float* __restrict__ d, int n) {
  int i = blockIdx.x * blockDim.x + threadIdx.x;
  if (i < n) d[i] = s[i];
}

// ---------------- batched input projection: xw[b,tc,:] = x[b,t0+tc,:] @ W + b ----------------
// A: bf16 [B, T, K] row-major; Wt: bf16 [2048, K] (pre-transposed); out: fp32 [B, TC, 2048]
__global__ void k_gemm_xw(const bhalf* __restrict__ A, const bhalf* __restrict__ Wt,
                          const float* __restrict__ bias, float* __restrict__ out,
                          int K, int T, int TC, int t0) {
  const int G = 2048;
  int lane = threadIdx.x & 31;
  int wave = threadIdx.x >> 5;                // 4 waves
  int r0 = blockIdx.x * 16;                   // row tile (rows = b*TC+tc)
  int cb = blockIdx.y * 64 + wave * 16;       // column tile base
  int row = r0 + (lane & 15);
  int b  = row / TC, tc = row % TC;
  const bhalf* arow = A  + ((size_t)(b * T + t0 + tc)) * K;
  const bhalf* brow = Wt + (size_t)(cb + (lane & 15)) * K;
  int koff8  = (lane >> 4) * 8;
  int koff16 = (lane >> 4) * 16;
  v8f acc = {};
  for (int k = 0; k < K; k += 32) {
    Frag fa, fb;
    fa.q[0] = *(const uint4*)(arow + k + koff8);
    fa.q[1] = *(const uint4*)(arow + k + koff8 + 16);
    fb.q[0] = *(const uint4*)(brow + k + koff8);
    fb.q[1] = *(const uint4*)(brow + k + koff8 + 16);
    acc = __builtin_amdgcn_wmma_f32_16x16x32_bf16(false, fa.v, false, fb.v,
                                                  (short)0, acc, false, false);
  }
  int n = cb + (lane & 15);
  float bv = bias[n];
  int mo = (lane >> 4) * 8;
  for (int r = 0; r < 8; ++r) {
    int gr = r0 + r + mo;                     // == b*TC+tc of this C row
    out[(size_t)gr * G + n] = acc[r] + bv;
  }
}

// ---------------- persistent recurrence over one chunk of TC steps ----------------
// grid = 32 blocks (one 16-column slice of H each), 256 threads (8 waves: 4 m-tiles x 2 gate-pairs).
// U slice (64 rows x 512) staged in LDS once; grid barrier (monotonic counter) between steps.
#define USTR 520   // padded LDS row stride (elements): 1040 B -> lanes hit distinct banks
__global__ void k_lstm_recur(const float* __restrict__ xw, int TC,
                             bhalf* hbA, bhalf* hbB,
                             const bhalf* __restrict__ Ut,
                             float* __restrict__ h, float* __restrict__ c,
                             bhalf* __restrict__ seqo, int t0, int T,
                             uint32_t* bar) {
  const int H = 512, G = 2048, B = 64, K = 512;
  __shared__ bhalf uts[64 * USTR];            // ~65 KB U slice (CDNA5: 320 KB LDS/WGP)
  __shared__ float zsh[4][64][16];            // 16 KB gate exchange
  int j0 = blockIdx.x * 16;

  // Stage this block's 64 U rows (4 gates x 16 columns) into LDS.
  for (int i = threadIdx.x; i < 64 * 64; i += blockDim.x) {   // 4096 uint4s
    int r = i >> 6, qc = i & 63;
    int grow = (r >> 4) * 512 + j0 + (r & 15);
    *(uint4*)&uts[r * USTR + qc * 8] = ((const uint4*)(Ut + (size_t)grow * K))[qc];
  }
  __syncthreads();

  int lane  = threadIdx.x & 31;
  int wave  = threadIdx.x >> 5;
  int mtile = wave & 3;
  int gbase = (wave >> 2) * 2;                // waves 0-3: gates i,f ; waves 4-7: gates g,o
  int m0 = mtile * 16;
  int j  = lane & 15;
  int koff8  = (lane >> 4) * 8;
  int koff16 = (lane >> 4) * 16;
  const bhalf* ub0 = &uts[(size_t)(gbase * 16 + j) * USTR];
  const bhalf* ub1 = &uts[(size_t)((gbase + 1) * 16 + j) * USTR];
  int mo = (lane >> 4) * 8;

  for (int tc = 0; tc < TC; ++tc) {
    int t = t0 + tc;
    const bhalf* hin  = (t & 1) ? hbB : hbA;  // t=0 reads zeroed hbA
    bhalf*       hout = (t & 1) ? hbA : hbB;
    const bhalf* arow = hin + (size_t)(m0 + j) * H;

    v8f acc0 = {}, acc1 = {};
    for (int k = 0; k < K; k += 32) {
      Frag fa, f0, f1;
      fa.q[0] = *(const uint4*)(arow + k + koff8);
      fa.q[1] = *(const uint4*)(arow + k + koff8 + 16);
      f0.q[0] = *(const uint4*)(ub0 + k + koff8);
      f0.q[1] = *(const uint4*)(ub0 + k + koff8 + 16);
      f1.q[0] = *(const uint4*)(ub1 + k + koff8);
      f1.q[1] = *(const uint4*)(ub1 + k + koff8 + 16);
      acc0 = __builtin_amdgcn_wmma_f32_16x16x32_bf16(false, fa.v, false, f0.v,
                                                     (short)0, acc0, false, false);
      acc1 = __builtin_amdgcn_wmma_f32_16x16x32_bf16(false, fa.v, false, f1.v,
                                                     (short)0, acc1, false, false);
    }
    for (int r = 0; r < 8; ++r) {
      int m = m0 + r + mo;                    // batch index
      size_t xrow = (size_t)(m * TC + tc) * G;
      zsh[gbase    ][m][j] = acc0[r] + xw[xrow + (size_t)(gbase    ) * 512 + j0 + j];
      zsh[gbase + 1][m][j] = acc1[r] + xw[xrow + (size_t)(gbase + 1) * 512 + j0 + j];
    }
    __syncthreads();
    for (int idx = threadIdx.x; idx < B * 16; idx += blockDim.x) {
      int bb = idx >> 4;
      int jj = idx & 15;
      float iv = sigm(zsh[0][bb][jj]);
      float fv = sigm(zsh[1][bb][jj]);
      float gv = tanhf(zsh[2][bb][jj]);
      float ov = sigm(zsh[3][bb][jj]);
      int col = j0 + jj;
      float cc = fv * c[bb * H + col] + iv * gv;
      float hh = ov * tanhf(cc);
      c[bb * H + col] = cc;
      h[bb * H + col] = hh;
      hout[bb * H + col] = f2bf(hh);
      seqo[((size_t)bb * T + t) * H + col] = f2bf(hh);
    }
    // ---- grid barrier: release our h writes, wait for all 32 blocks ----
    if (tc != TC - 1) {
      __threadfence();
      __syncthreads();
      if (threadIdx.x == 0) {
        __hip_atomic_fetch_add(bar, 1u, __ATOMIC_RELEASE, __HIP_MEMORY_SCOPE_AGENT);
        uint32_t target = 32u * (uint32_t)(tc + 1);
        while (__hip_atomic_load(bar, __ATOMIC_ACQUIRE, __HIP_MEMORY_SCOPE_AGENT) < target)
          __builtin_amdgcn_s_sleep(1);
      }
      __syncthreads();
    }
  }
}

// ---------------- host orchestration ----------------
extern "C" void kernel_launch(void* const* d_in, const int* in_sizes, int n_in,
                              void* d_out, int out_size, void* d_ws, size_t ws_size,
                              hipStream_t stream) {
  (void)in_sizes; (void)n_in; (void)out_size; (void)ws_size;
  const int B = 64, T = 512, D = 256, H = 512, G = 2048, TC = 64;
  const float* x       = (const float*)d_in[0];
  const float* W[3]    = {(const float*)d_in[1], (const float*)d_in[4], (const float*)d_in[7]};
  const float* U[3]    = {(const float*)d_in[2], (const float*)d_in[5], (const float*)d_in[8]};
  const float* bias[3] = {(const float*)d_in[3], (const float*)d_in[6], (const float*)d_in[9]};
  float* out = (float*)d_out;

  char* base = (char*)d_ws; size_t off = 0;
  auto wsalloc = [&](size_t bytes) -> char* {
    char* r = base + off; off = (off + bytes + 255) & ~(size_t)255; return r;
  };
  bhalf* xb0  = (bhalf*)wsalloc((size_t)B * T * D * 2);
  bhalf* seqA = (bhalf*)wsalloc((size_t)B * T * H * 2);
  bhalf* seqB = (bhalf*)wsalloc((size_t)B * T * H * 2);
  bhalf* Wt[3]; bhalf* Ut[3];
  Wt[0] = (bhalf*)wsalloc((size_t)G * D * 2);
  Ut[0] = (bhalf*)wsalloc((size_t)G * H * 2);
  Wt[1] = (bhalf*)wsalloc((size_t)G * H * 2);
  Ut[1] = (bhalf*)wsalloc((size_t)G * H * 2);
  Wt[2] = (bhalf*)wsalloc((size_t)G * H * 2);
  Ut[2] = (bhalf*)wsalloc((size_t)G * H * 2);
  float* xw   = (float*)wsalloc((size_t)B * TC * G * 4);
  float* hbuf = (float*)wsalloc((size_t)B * H * 4);
  float* cbuf = (float*)wsalloc((size_t)B * H * 4);
  bhalf* hbA  = (bhalf*)wsalloc((size_t)B * H * 2);
  bhalf* hbB  = (bhalf*)wsalloc((size_t)B * H * 2);
  uint32_t* bar = (uint32_t*)wsalloc(256);

  // One-time precision/layout conversion (bf16, weights transposed to [4H,K])
  int nx = B * T * D;
  k_cvt_bf16<<<(nx + 255) / 256, 256, 0, stream>>>(x, xb0, nx);
  k_transpose_bf16<<<(D * G + 255) / 256, 256, 0, stream>>>(W[0], Wt[0], D, G);
  k_transpose_bf16<<<(H * G + 255) / 256, 256, 0, stream>>>(U[0], Ut[0], H, G);
  k_transpose_bf16<<<(H * G + 255) / 256, 256, 0, stream>>>(W[1], Wt[1], H, G);
  k_transpose_bf16<<<(H * G + 255) / 256, 256, 0, stream>>>(U[1], Ut[1], H, G);
  k_transpose_bf16<<<(H * G + 255) / 256, 256, 0, stream>>>(W[2], Wt[2], H, G);
  k_transpose_bf16<<<(H * G + 255) / 256, 256, 0, stream>>>(U[2], Ut[2], H, G);

  const bhalf* lin[3]  = {xb0, seqA, seqB};
  bhalf*       lout[3] = {seqA, seqB, seqA};   // layer2 seq is dead, reuse seqA
  int Kin[3] = {D, H, H};

  for (int L = 0; L < 3; ++L) {
    k_zero_u32<<<(B * H + 255) / 256, 256, 0, stream>>>((uint32_t*)cbuf, B * H);
    k_zero_u32<<<(B * H / 2 + 255) / 256, 256, 0, stream>>>((uint32_t*)hbA, B * H / 2);
    for (int t0 = 0; t0 < T; t0 += TC) {
      dim3 gg(B * TC / 16, G / 64);
      k_gemm_xw<<<gg, 128, 0, stream>>>(lin[L], Wt[L], bias[L], xw, Kin[L], T, TC, t0);
      k_zero_u32<<<1, 64, 0, stream>>>(bar, 1);
      k_lstm_recur<<<H / 16, 256, 0, stream>>>(xw, TC, hbA, hbB, Ut[L],
                                               hbuf, cbuf, lout[L], t0, T, bar);
    }
    int n = B * H;
    float* oh = out + (size_t)(1 + 2 * L) * n;   // slots: h0,c0,h1,c1,h2,c2
    float* oc = out + (size_t)(2 + 2 * L) * n;
    k_copy_f32<<<(n + 255) / 256, 256, 0, stream>>>(hbuf, oh, n);
    k_copy_f32<<<(n + 255) / 256, 256, 0, stream>>>(cbuf, oc, n);
    if (L == 2) k_copy_f32<<<(n + 255) / 256, 256, 0, stream>>>(hbuf, out, n);  // out == h2
  }
}
